// GPT_72722386256487
// MI455X (gfx1250) — matrix operand order repaired and checked by reading, not compiled, not run on hardware
//
#include <hip/hip_runtime.h>
#include <hip/hip_bf16.h>
#include <math.h>

// ---------------------------------------------------------------------------
// Types for CDNA5 WMMA (wave32): v_wmma_f32_16x16x32_bf16
// ---------------------------------------------------------------------------
typedef __bf16 v16bf  __attribute__((ext_vector_type(16)));
typedef float  v8f    __attribute__((ext_vector_type(8)));
typedef __bf16 bf16x8 __attribute__((ext_vector_type(8)));

#define WMMA_BF16(a, b, c) \
    __builtin_amdgcn_wmma_f32_16x16x32_bf16(false, (a), false, (b), (short)0, (c), false, false)

// Model constants (match reference setup_inputs)
constexpr int cB = 2, cT = 2048, cD = 512, cH = 8, cHD = 64, cF = 2048, cL = 6, cV = 32000;
constexpr int cN = cB * cT; // 4096 rows in the flattened [B*T, D] activation

static __device__ __forceinline__ v8f v8f_zero() {
    v8f z;
#pragma unroll
    for (int i = 0; i < 8; ++i) z[i] = 0.0f;
    return z;
}

// Load a 16-element bf16 fragment as two contiguous 8-element (16B) chunks.
static __device__ __forceinline__ v16bf load_frag16(const __bf16* rowbase, int lo, int hi) {
    bf16x8 l8 = *(const bf16x8*)(rowbase + lo);
    bf16x8 h8 = *(const bf16x8*)(rowbase + hi);
    v16bf a;
#pragma unroll
    for (int i = 0; i < 8; ++i) { a[i] = l8[i]; a[i + 8] = h8[i]; }
    return a;
}

// ---------------------------------------------------------------------------
// fp32 -> bf16 conversion (weights), grid-stride
// ---------------------------------------------------------------------------
__global__ void cvt_f32_to_bf16_kernel(const float* __restrict__ in, __bf16* __restrict__ out, int n) {
    for (int i = blockIdx.x * blockDim.x + threadIdx.x; i < n; i += gridDim.x * blockDim.x)
        out[i] = (__bf16)in[i];
}

// ---------------------------------------------------------------------------
// wte [V, D] fp32  ->  wteT [D, V] bf16 (for tied logits head as row-major B)
// ---------------------------------------------------------------------------
__global__ __launch_bounds__(256) void transpose_wte_kernel(const float* __restrict__ wte,
                                                            __bf16* __restrict__ wteT) {
    __shared__ float tile[32][33];
    const int tx = threadIdx.x, ty = threadIdx.y; // block (32, 8)
    const int v0 = blockIdx.x * 32, d0 = blockIdx.y * 32;
#pragma unroll
    for (int j = 0; j < 4; ++j)
        tile[ty + j * 8][tx] = wte[(size_t)(v0 + ty + j * 8) * cD + d0 + tx];
    __syncthreads();
#pragma unroll
    for (int j = 0; j < 4; ++j)
        wteT[(size_t)(d0 + ty + j * 8) * cV + v0 + tx] = (__bf16)tile[tx][ty + j * 8];
}

// ---------------------------------------------------------------------------
// x[n, :] = wte[idx[n], :] + wpe[n % T, :]   (fp32 residual stream)
// ---------------------------------------------------------------------------
__global__ __launch_bounds__(256) void embed_kernel(const int* __restrict__ idx,
                                                    const float* __restrict__ wte,
                                                    const float* __restrict__ wpe,
                                                    float* __restrict__ x) {
    const int n = blockIdx.x;
    const int t = n & (cT - 1);
    const int tok = idx[n];
#pragma unroll
    for (int j = 0; j < 2; ++j) {
        const int d = threadIdx.x + j * 256;
        x[(size_t)n * cD + d] = wte[(size_t)tok * cD + d] + wpe[(size_t)t * cD + d];
    }
}

// ---------------------------------------------------------------------------
// LayerNorm over D=512, fp32 in -> bf16 out. One block (256 thr) per row.
// ---------------------------------------------------------------------------
__global__ __launch_bounds__(256) void layernorm_kernel(const float* __restrict__ x,
                                                        const float* __restrict__ g,
                                                        const float* __restrict__ bta,
                                                        __bf16* __restrict__ out) {
    __shared__ float sdata[256];
    const int row = blockIdx.x;
    const int tid = threadIdx.x;
    const float* xr = x + (size_t)row * cD;
    const float a0 = xr[tid], a1 = xr[tid + 256];
    sdata[tid] = a0 + a1;
    __syncthreads();
    for (int off = 128; off; off >>= 1) {
        if (tid < off) sdata[tid] += sdata[tid + off];
        __syncthreads();
    }
    const float mu = sdata[0] * (1.0f / (float)cD);
    __syncthreads();
    const float d0 = a0 - mu, d1 = a1 - mu;
    sdata[tid] = d0 * d0 + d1 * d1;
    __syncthreads();
    for (int off = 128; off; off >>= 1) {
        if (tid < off) sdata[tid] += sdata[tid + off];
        __syncthreads();
    }
    const float rstd = rsqrtf(sdata[0] * (1.0f / (float)cD) + 1e-5f);
    out[(size_t)row * cD + tid]       = (__bf16)(d0 * rstd * g[tid] + bta[tid]);
    out[(size_t)row * cD + tid + 256] = (__bf16)(d1 * rstd * g[tid + 256] + bta[tid + 256]);
}

// ---------------------------------------------------------------------------
// Generic WMMA GEMM:  C[M, Nc] = A[M, K](bf16) x B[K, Nc](bf16)  (+ epilogue)
//   EPI 0: + bias, store bf16
//   EPI 1: + bias, exact GELU, store bf16
//   EPI 2: + bias + residual, store fp32 (resid may alias outp)
//   EPI 3: store fp32 (logits, no bias)
// Block tile 64x256, BK=32. 8 waves in a 2x4 grid, each owning 32x64
// (2x4 wmma accumulators). A tile staged via GLOBAL_LOAD_ASYNC_TO_LDS_B128
// (ASYNCcnt); B tile transposed to [n][k] in LDS so fragment loads are b128.
// ---------------------------------------------------------------------------
template <int EPI>
__global__ __launch_bounds__(256) void gemm_bf16_kernel(const __bf16* __restrict__ A,
                                                        const __bf16* __restrict__ Bw,
                                                        const float* __restrict__ bias,
                                                        const float* resid,
                                                        void* __restrict__ outp,
                                                        int K, int Nc) {
    constexpr int BM = 64, BN = 256, BK = 32, PAD = 8;
    __shared__ __align__(16) __bf16 As[BM][BK + PAD]; // [m][k]
    __shared__ __align__(16) __bf16 Bt[BN][BK + PAD]; // [n][k] (transposed stage)

    const int tid  = threadIdx.x;
    const int lane = tid & 31, wid = tid >> 5;
    const int r = lane & 15, hf = lane >> 4;
    const int wm = (wid >> 2) * 32, wn = (wid & 3) * 64;
    const size_t m0 = (size_t)blockIdx.y * BM;
    const size_t n0 = (size_t)blockIdx.x * BN;
    const int arow = tid >> 2, acol = (tid & 3) * 8;   // A tile: 64x32, 16B per thread
    const int brow = tid >> 3, bcol = (tid & 7) * 32;  // B tile: 32x256, 64B per thread

    // LDS byte address of this thread's A-tile slot (low 32 bits of the
    // generic shared pointer are the wave-relative LDS offset).
    const unsigned a_lds = (unsigned)(uintptr_t)&As[arow][acol];
    const unsigned long long a_base = (unsigned long long)(uintptr_t)A;

    v8f acc[2][4];
#pragma unroll
    for (int i = 0; i < 2; ++i)
#pragma unroll
        for (int j = 0; j < 4; ++j) acc[i][j] = v8f_zero();

    for (int k0 = 0; k0 < K; k0 += BK) {
        __syncthreads(); // WAR vs previous iteration's fragment reads

        // --- A tile: async DMA global -> LDS (16B per lane, ASYNCcnt) ---
        {
            const unsigned a_voff = (unsigned)(((m0 + arow) * (size_t)K + k0 + acol) * 2u);
            asm volatile("global_load_async_to_lds_b128 %0, %1, %2"
                         :
                         : "v"(a_lds), "v"(a_voff), "s"(a_base)
                         : "memory");
        }

        // --- B tile: load 64B per thread, transpose-store into Bt[n][k] ---
        {
            const __bf16* bp = &Bw[(size_t)(k0 + brow) * Nc + n0 + bcol];
#pragma unroll
            for (int c = 0; c < 4; ++c) {
                bf16x8 bv = *(const bf16x8*)(bp + c * 8);
#pragma unroll
                for (int j = 0; j < 8; ++j) Bt[bcol + c * 8 + j][brow] = bv[j];
            }
        }
        if (k0 + BK < K) // emits global_prefetch_b8 for next K-tile of B
            __builtin_prefetch(&Bw[(size_t)(k0 + BK + brow) * Nc + n0 + bcol], 0, 1);

        asm volatile("s_wait_asynccnt 0x0" ::: "memory");
        __syncthreads();

        v16bf af[2], bfr[4];
#pragma unroll
        for (int i = 0; i < 2; ++i) // A: lane = row, K contiguous 16 per half
            af[i] = load_frag16(&As[wm + i * 16 + r][0], hf * 16, hf * 16 + 8);
#pragma unroll
        for (int j = 0; j < 4; ++j) // B: lane = col, K contiguous 16 per half
            bfr[j] = load_frag16(&Bt[wn + j * 16 + r][0], hf * 16, hf * 16 + 8);
#pragma unroll
        for (int i = 0; i < 2; ++i)
#pragma unroll
            for (int j = 0; j < 4; ++j)
                acc[i][j] = WMMA_BF16(af[i], bfr[j], acc[i][j]);
    }

    // C layout: elem e -> row = e + hf*8, col = r (per 16x16 tile)
#pragma unroll
    for (int i = 0; i < 2; ++i) {
#pragma unroll
        for (int j = 0; j < 4; ++j) {
#pragma unroll
            for (int e = 0; e < 8; ++e) {
                const size_t row = m0 + wm + i * 16 + e + hf * 8;
                const size_t col = n0 + wn + j * 16 + r;
                float v = acc[i][j][e];
                if constexpr (EPI == 0 || EPI == 1 || EPI == 2) v += bias[col];
                if constexpr (EPI == 1) v = 0.5f * v * (1.0f + erff(v * 0.70710678118654752f));
                const size_t o = row * (size_t)Nc + col;
                if constexpr (EPI == 0 || EPI == 1) ((__bf16*)outp)[o] = (__bf16)v;
                else if constexpr (EPI == 2)        ((float*)outp)[o]  = resid[o] + v;
                else                                ((float*)outp)[o]  = v;
            }
        }
    }
}

// ---------------------------------------------------------------------------
// Flash attention: one wave per (b, h, 16-query tile). qkv is [N, 3D] bf16
// produced by the qkv GEMM (q | k | v column blocks); out is [N, D] bf16.
// ---------------------------------------------------------------------------
__global__ __launch_bounds__(32) void attention_kernel(const __bf16* __restrict__ qkv,
                                                       __bf16* __restrict__ out) {
    __shared__ __align__(16) __bf16 Kt[32][72]; // [key][d], 32 keys x 64 dims (+pad)
    __shared__ __align__(16) __bf16 Vt[32][72];
    __shared__ __align__(16) __bf16 Ps[16][40]; // P tile bounce: C-layout -> A-layout

    const int lane = threadIdx.x;
    const int r = lane & 15, hf = lane >> 4;
    const int qt = blockIdx.x, h = blockIdx.y, b = blockIdx.z;
    const int q0 = qt * 16;

    // Q fragments: A-matrix [16 queries x 64 dims], two K-chunks of 32
    const __bf16* qp = qkv + ((size_t)(b * cT + q0 + r)) * (3 * cD) + h * cHD;
    v16bf aq[2];
#pragma unroll
    for (int kc = 0; kc < 2; ++kc)
        aq[kc] = load_frag16(qp, kc * 32 + hf * 16, kc * 32 + hf * 16 + 8);

    float m_run[8], l_run[8];
    v8f acc[4]; // 16 queries x 64 dims output accumulator
#pragma unroll
    for (int e = 0; e < 8; ++e) { m_run[e] = -1e30f; l_run[e] = 0.0f; }
#pragma unroll
    for (int j = 0; j < 4; ++j) acc[j] = v8f_zero();

    const int ntiles = (q0 + 16 + 31) >> 5; // causal: keys 0 .. q0+15
    for (int kt = 0; kt < ntiles; ++kt) {
        const int kbase = kt * 32;
        __syncthreads(); // WAR on Kt/Vt vs previous iteration
        // Stage K,V tiles: lane loads one 64-dim row (128B) each, coalesced
        const __bf16* kp = qkv + ((size_t)(b * cT + kbase + lane)) * (3 * cD) + cD + h * cHD;
        const __bf16* vp = kp + cD;
#pragma unroll
        for (int j = 0; j < 8; ++j) {
            *(bf16x8*)&Kt[lane][j * 8] = *(const bf16x8*)&kp[j * 8];
            *(bf16x8*)&Vt[lane][j * 8] = *(const bf16x8*)&vp[j * 8];
        }
        __syncthreads();

        // Scores: S = Q x K^T  (B-frag: lane = key col, elems = contiguous d)
        v8f s[2];
#pragma unroll
        for (int nf = 0; nf < 2; ++nf) {
            v8f z = v8f_zero();
            v16bf kf0 = load_frag16(&Kt[nf * 16 + r][0], hf * 16, hf * 16 + 8);
            v16bf kf1 = load_frag16(&Kt[nf * 16 + r][0], 32 + hf * 16, 32 + hf * 16 + 8);
            z = WMMA_BF16(aq[0], kf0, z);
            z = WMMA_BF16(aq[1], kf1, z);
            s[nf] = z;
        }

        // Online softmax (fp32). Row lives across the 16 lanes of a half.
        float p0a[8], p1a[8];
#pragma unroll
        for (int e = 0; e < 8; ++e) {
            const int qidx = q0 + e + hf * 8;
            float v0 = s[0][e] * 0.125f; // 1/sqrt(64)
            float v1 = s[1][e] * 0.125f;
            if (kbase + r > qidx)      v0 = -1e30f;
            if (kbase + 16 + r > qidx) v1 = -1e30f;
            float mx = fmaxf(v0, v1);
#pragma unroll
            for (int msk = 1; msk < 16; msk <<= 1) mx = fmaxf(mx, __shfl_xor(mx, msk, 32));
            const float mnew  = fmaxf(m_run[e], mx);
            const float alpha = __expf(m_run[e] - mnew);
            const float p0 = __expf(v0 - mnew);
            const float p1 = __expf(v1 - mnew);
            float rs = p0 + p1;
#pragma unroll
            for (int msk = 1; msk < 16; msk <<= 1) rs += __shfl_xor(rs, msk, 32);
            l_run[e] = l_run[e] * alpha + rs;
            m_run[e] = mnew;
#pragma unroll
            for (int j = 0; j < 4; ++j) acc[j][e] *= alpha;
            p0a[e] = p0; p1a[e] = p1;
        }

        // Bounce P through LDS to convert C-layout -> A-layout
        __syncthreads(); // WAR vs previous iteration's Ps reads
#pragma unroll
        for (int e = 0; e < 8; ++e) {
            Ps[e + hf * 8][r]      = (__bf16)p0a[e];
            Ps[e + hf * 8][16 + r] = (__bf16)p1a[e];
        }
        __syncthreads();
        v16bf pf = load_frag16(&Ps[r][0], hf * 16, hf * 16 + 8);

        // acc += P x V   (V B-frag: lane = d col, elems = key rows, gathered)
#pragma unroll
        for (int jd = 0; jd < 4; ++jd) {
            v16bf vf;
#pragma unroll
            for (int i = 0; i < 16; ++i) vf[i] = Vt[hf * 16 + i][jd * 16 + r];
            acc[jd] = WMMA_BF16(pf, vf, acc[jd]);
        }
    }

    // Normalize and store [16 x 64] tile
#pragma unroll
    for (int jd = 0; jd < 4; ++jd)
#pragma unroll
        for (int e = 0; e < 8; ++e) {
            const int q = q0 + e + hf * 8;
            const int d = jd * 16 + r;
            out[((size_t)(b * cT + q)) * cD + h * cHD + d] = (__bf16)(acc[jd][e] / l_run[e]);
        }
}

// ---------------------------------------------------------------------------
// Host orchestration
// ---------------------------------------------------------------------------
extern "C" void kernel_launch(void* const* d_in, const int* in_sizes, int n_in,
                              void* d_out, int out_size, void* d_ws, size_t ws_size,
                              hipStream_t stream) {
    (void)in_sizes; (void)n_in; (void)out_size; (void)ws_size;

    const int*   idx    = (const int*)  d_in[0];
    const float* wte    = (const float*)d_in[1];
    const float* wpe    = (const float*)d_in[2];
    const float* ln1_g  = (const float*)d_in[3];
    const float* ln1_b  = (const float*)d_in[4];
    const float* qkv_w  = (const float*)d_in[5];
    const float* qkv_b  = (const float*)d_in[6];
    const float* proj_w = (const float*)d_in[7];
    const float* proj_b = (const float*)d_in[8];
    const float* ln2_g  = (const float*)d_in[9];
    const float* ln2_b  = (const float*)d_in[10];
    const float* fc1_w  = (const float*)d_in[11];
    const float* fc1_b  = (const float*)d_in[12];
    const float* fc2_w  = (const float*)d_in[13];
    const float* fc2_b  = (const float*)d_in[14];
    const float* lnf_g  = (const float*)d_in[15];
    const float* lnf_b  = (const float*)d_in[16];

    char* ws = (char*)d_ws;
    size_t off = 0;
    auto alloc = [&](size_t bytes) -> void* {
        void* p = ws + off;
        off += (bytes + 255) & ~(size_t)255;
        return p;
    };
    float*  x      = (float*) alloc((size_t)cN * cD * 4);       // fp32 residual stream
    __bf16* hbf    = (__bf16*)alloc((size_t)cN * cD * 2);       // LN output
    __bf16* qkvbf  = (__bf16*)alloc((size_t)cN * 3 * cD * 2);   // qkv
    __bf16* attbf  = (__bf16*)alloc((size_t)cN * cD * 2);       // attention out
    __bf16* ffnbf  = (__bf16*)alloc((size_t)cN * cF * 2);       // gelu(fc1)
    __bf16* wq_bf  = (__bf16*)alloc((size_t)cL * cD * 3 * cD * 2);
    __bf16* wp_bf  = (__bf16*)alloc((size_t)cL * cD * cD * 2);
    __bf16* w1_bf  = (__bf16*)alloc((size_t)cL * cD * cF * 2);
    __bf16* w2_bf  = (__bf16*)alloc((size_t)cL * cF * cD * 2);
    __bf16* wteT   = (__bf16*)alloc((size_t)cD * cV * 2);

    // Weight precision conversion (every call; no cross-call state)
    cvt_f32_to_bf16_kernel<<<4096, 256, 0, stream>>>(qkv_w,  wq_bf, cL * cD * 3 * cD);
    cvt_f32_to_bf16_kernel<<<4096, 256, 0, stream>>>(proj_w, wp_bf, cL * cD * cD);
    cvt_f32_to_bf16_kernel<<<4096, 256, 0, stream>>>(fc1_w,  w1_bf, cL * cD * cF);
    cvt_f32_to_bf16_kernel<<<4096, 256, 0, stream>>>(fc2_w,  w2_bf, cL * cF * cD);
    transpose_wte_kernel<<<dim3(cV / 32, cD / 32), dim3(32, 8), 0, stream>>>(wte, wteT);

    // Embedding
    embed_kernel<<<cN, 256, 0, stream>>>(idx, wte, wpe, x);

    // Transformer blocks
    for (int l = 0; l < cL; ++l) {
        layernorm_kernel<<<cN, 256, 0, stream>>>(x, ln1_g + l * cD, ln1_b + l * cD, hbf);
        gemm_bf16_kernel<0><<<dim3(3 * cD / 256, cN / 64), 256, 0, stream>>>(
            hbf, wq_bf + (size_t)l * cD * 3 * cD, qkv_b + l * 3 * cD, nullptr, qkvbf, cD, 3 * cD);
        attention_kernel<<<dim3(cT / 16, cH, cB), 32, 0, stream>>>(qkvbf, attbf);
        gemm_bf16_kernel<2><<<dim3(cD / 256, cN / 64), 256, 0, stream>>>(
            attbf, wp_bf + (size_t)l * cD * cD, proj_b + l * cD, x, x, cD, cD);
        layernorm_kernel<<<cN, 256, 0, stream>>>(x, ln2_g + l * cD, ln2_b + l * cD, hbf);
        gemm_bf16_kernel<1><<<dim3(cF / 256, cN / 64), 256, 0, stream>>>(
            hbf, w1_bf + (size_t)l * cD * cF, fc1_b + l * cF, nullptr, ffnbf, cD, cF);
        gemm_bf16_kernel<2><<<dim3(cD / 256, cN / 64), 256, 0, stream>>>(
            ffnbf, w2_bf + (size_t)l * cF * cD, fc2_b + l * cD, x, x, cF, cD);
    }

    // Final LN + tied logits head (fp32 out)
    layernorm_kernel<<<cN, 256, 0, stream>>>(x, lnf_g, lnf_b, hbf);
    gemm_bf16_kernel<3><<<dim3(cV / 256, cN / 64), 256, 0, stream>>>(
        hbf, wteT, nullptr, nullptr, d_out, cD, cV);
}
